// minGRU_42743514529993
// MI455X (gfx1250) — matrix operand order repaired and checked by reading, not compiled, not run on hardware
//
#include <hip/hip_runtime.h>

// ---------------------------------------------------------------------------
// minGRU (L=8192, N=2048): fused dual bf16-WMMA GEMM (Wz & Wh share the A
// fragment), K-slab tiles staged through LDS with CDNA5 async-to-LDS DMA in a
// double-buffered pipeline (ASYNCcnt), sigmoid-gate epilogue with reset-flag
// folding, then a 3-phase chunked linear scan over L.
// ---------------------------------------------------------------------------

typedef __attribute__((ext_vector_type(16))) __bf16 v16bf;
typedef __attribute__((ext_vector_type(8)))  float  v8f;
typedef __attribute__((ext_vector_type(4)))  int    v4i;
typedef __attribute__((address_space(1))) v4i gv4i;   // global v4i
typedef __attribute__((address_space(3))) v4i lv4i;   // LDS v4i

#define KSTEP  32
#define LROW   40                      // padded LDS row stride in shorts (80 B)
#define A_ROWS 64
#define B_ROWS 128
#define OFF_A  0
#define OFF_Z  (A_ROWS * LROW)         // 2560 shorts
#define OFF_H  (OFF_Z + B_ROWS * LROW) // 7680 shorts
#define STAGE  (OFF_H + B_ROWS * LROW) // 12800 shorts = 25600 B per stage

union Frag { uint4 q[2]; v16bf v; };

// ---- async 16-byte global->LDS copy (ASYNCcnt) -----------------------------
__device__ __forceinline__ void async_copy16(const unsigned short* g,
                                             unsigned short* l) {
#if defined(__has_builtin) && __has_builtin(__builtin_amdgcn_global_load_async_to_lds_b128)
  // builtin signature: (v4i as1* src, v4i as3* dst, imm offset, imm cpol)
  // low 32 flat bits of an LDS generic pointer == wave-relative LDS address
  __builtin_amdgcn_global_load_async_to_lds_b128(
      (gv4i*)(size_t)g, (lv4i*)(unsigned int)(size_t)l, 0, 0);
#else
  unsigned int lds = (unsigned int)(size_t)l;
  asm volatile("global_load_async_to_lds_b128 %0, %1, off"
               :: "v"(lds), "v"(g) : "memory");
#endif
}

__device__ __forceinline__ void wait_async0() {
#if defined(__has_builtin) && __has_builtin(__builtin_amdgcn_s_wait_asynccnt)
  __builtin_amdgcn_s_wait_asynccnt(0);
#else
  asm volatile("s_wait_asynccnt 0x0" ::: "memory");
#endif
}

// Issue the 1280 async 16-B chunks for one K-slab stage (5 per thread).
__device__ __forceinline__ void stage_issue(const unsigned short* __restrict__ ubf,
                                            const unsigned short* __restrict__ wzbf,
                                            const unsigned short* __restrict__ whbf,
                                            unsigned short* st,
                                            int Lblk, int Mblk, int kk, int N,
                                            int tid) {
  {
    const int row = tid >> 2, q = tid & 3;            // A: 64 rows x 4 chunks
    async_copy16(ubf + (size_t)(Lblk + row) * N + kk + q * 8,
                 st + OFF_A + row * LROW + q * 8);
  }
#pragma unroll
  for (int i = 0; i < 2; ++i) {                       // Wz: 128 rows x 4 chunks
    const int cc = tid + i * 256;
    const int row = cc >> 2, q = cc & 3;
    async_copy16(wzbf + (size_t)(Mblk + row) * N + kk + q * 8,
                 st + OFF_Z + row * LROW + q * 8);
  }
#pragma unroll
  for (int i = 0; i < 2; ++i) {                       // Wh: 128 rows x 4 chunks
    const int cc = tid + i * 256;
    const int row = cc >> 2, q = cc & 3;
    async_copy16(whbf + (size_t)(Mblk + row) * N + kk + q * 8,
                 st + OFF_H + row * LROW + q * 8);
  }
}

// ---- f32 -> bf16 (round to nearest even) conversion ------------------------
__global__ void cvt_bf16_kernel(const float* __restrict__ src,
                                unsigned short* __restrict__ dst, size_t n) {
  size_t i = (size_t)blockIdx.x * blockDim.x + threadIdx.x;
  size_t stride = (size_t)gridDim.x * blockDim.x;
  for (; i < n; i += stride) {
    unsigned u = __float_as_uint(src[i]);
    u += 0x7FFFu + ((u >> 16) & 1u);
    dst[i] = (unsigned short)(u >> 16);
  }
}

// ---- fused dual GEMM + gate epilogue ---------------------------------------
// Block: 256 threads = 8 waves, wave tile 16(L) x 64(M), WG tile 64(L) x 128(M).
// Double-buffered LDS stages; per K-step each wave loads all 9 fragments into
// distinct registers first, then runs the 8 WMMAs as an uninterrupted chain.
__global__ __launch_bounds__(256)
void gemm_gate_kernel(const unsigned short* __restrict__ ubf,
                      const unsigned short* __restrict__ wzbf,
                      const unsigned short* __restrict__ whbf,
                      const float* __restrict__ bz,
                      const float* __restrict__ bh,
                      const unsigned char* __restrict__ dflag,
                      float* __restrict__ aArr,   // a_t = d ? 0 : (1 - sigma)
                      float* __restrict__ bArr,   // b_t = sigma * h_sim
                      int N) {
  __shared__ unsigned short smem[2 * STAGE];      // 50 KB

  const int tid  = threadIdx.x;
  const int lane = tid & 31;
  const int wave = tid >> 5;
  const int lo = lane & 15;       // A: row ; B: column ; D: column
  const int hi = lane >> 4;       // K-half selector / D row-half selector
  const int wl = wave & 3;        // 4 L-subtiles of 16
  const int wm = wave >> 2;       // 2 M-subtiles of 64
  const int Lblk = blockIdx.y * 64;
  const int Mblk = blockIdx.x * 128;

  v8f accZ[4] = {{0.f}}, accH[4] = {{0.f}};

  // Preload stage 0, then pipeline: prefetch k+1 while computing k.
  stage_issue(ubf, wzbf, whbf, smem, Lblk, Mblk, 0, N, tid);
  wait_async0();
  __syncthreads();

  int cur = 0;
  for (int kk = 0; kk < N; kk += KSTEP) {
    unsigned short* st  = smem + cur * STAGE;
    unsigned short* nxt = smem + (cur ^ 1) * STAGE;
    if (kk + KSTEP < N)
      stage_issue(ubf, wzbf, whbf, nxt, Lblk, Mblk, kk + KSTEP, N, tid);

    // Load ALL fragments first (distinct registers) ...
    Frag af, bzf[4], bhf[4];
    // A fragment (16x32 bf16): lanes 0-15 K {0..15}, 16-31 K {16..31}
    const unsigned short* pa = st + OFF_A + (wl * 16 + lo) * LROW + hi * 16;
    af.q[0] = *(const uint4*)(pa);
    af.q[1] = *(const uint4*)(pa + 8);
#pragma unroll
    for (int j = 0; j < 4; ++j) {
      const int brow = (wm * 64 + j * 16 + lo) * LROW + hi * 16;
      const unsigned short* pz = st + OFF_Z + brow;
      const unsigned short* ph = st + OFF_H + brow;
      bzf[j].q[0] = *(const uint4*)(pz);
      bzf[j].q[1] = *(const uint4*)(pz + 8);
      bhf[j].q[0] = *(const uint4*)(ph);
      bhf[j].q[1] = *(const uint4*)(ph + 8);
    }

    // ... then issue the 8 WMMAs back-to-back (single dscnt wait).
#pragma unroll
    for (int j = 0; j < 4; ++j) {
      accZ[j] = __builtin_amdgcn_wmma_f32_16x16x32_bf16(
          false, af.v, false, bzf[j].v, (short)0, accZ[j], false, false);
      accH[j] = __builtin_amdgcn_wmma_f32_16x16x32_bf16(
          false, af.v, false, bhf[j].v, (short)0, accH[j], false, false);
    }

    wait_async0();     // my DMA into nxt is complete
    __syncthreads();   // everyone done reading cur + nxt fully populated
    cur ^= 1;
  }

  // Epilogue: D layout -> row = hi*8 + r, col = lo.
#pragma unroll
  for (int j = 0; j < 4; ++j) {
    const int m = Mblk + wm * 64 + j * 16 + lo;
    const float bzm = bz[m], bhm = bh[m];
#pragma unroll
    for (int r = 0; r < 8; ++r) {
      const int lr = Lblk + wl * 16 + hi * 8 + r;
      const float zlin = accZ[j][r] + bzm;
      const float hlin = accH[j][r] + bhm;
      const float zg = 1.0f / (1.0f + __expf(-zlin));
      const float av = dflag[lr] ? 0.0f : (1.0f - zg);
      const float bv = zg * hlin;
      const size_t o = (size_t)lr * N + m;
      aArr[o] = av;
      bArr[o] = bv;
    }
  }
}

// ---- phase 1: per-chunk composite (A_c, B_c) -------------------------------
__global__ __launch_bounds__(256)
void chunk_reduce_kernel(const float* __restrict__ aArr,
                         const float* __restrict__ bArr,
                         float* __restrict__ Ac, float* __restrict__ Bc,
                         int N, int T) {
  const int n = blockIdx.x * blockDim.x + threadIdx.x;
  const int c = blockIdx.y;
  const size_t base = (size_t)c * T * N + n;
  float A = 1.0f, B = 0.0f;
  for (int t = 0; t < T; ++t) {
    const float av = aArr[base + (size_t)t * N];
    const float bv = bArr[base + (size_t)t * N];
    B = av * B + bv;
    A = av * A;
  }
  Ac[(size_t)c * N + n] = A;
  Bc[(size_t)c * N + n] = B;
}

// ---- phase 2: sequential carry combine across chunks (parallel over n) -----
__global__ __launch_bounds__(256)
void carry_scan_kernel(const float* __restrict__ Ac, const float* __restrict__ Bc,
                       const float* __restrict__ x, float* __restrict__ Hc,
                       int N, int C) {
  const int n = blockIdx.x * blockDim.x + threadIdx.x;
  float h = x[n];                       // h_{-1} = x
  for (int c = 0; c < C; ++c) {
    Hc[(size_t)c * N + n] = h;          // carry entering chunk c
    h = Ac[(size_t)c * N + n] * h + Bc[(size_t)c * N + n];
  }
}

// ---- phase 3: apply carries, emit hs[1:] and hs[-1] ------------------------
__global__ __launch_bounds__(256)
void chunk_apply_kernel(const float* __restrict__ aArr,
                        const float* __restrict__ bArr,
                        const float* __restrict__ Hc,
                        float* __restrict__ out, int N, int T, int L) {
  const int n = blockIdx.x * blockDim.x + threadIdx.x;
  const int c = blockIdx.y;
  float h = Hc[(size_t)c * N + n];
  const int t0 = c * T;
  for (int t = 0; t < T; ++t) {
    const size_t o = (size_t)(t0 + t) * N + n;
    h = aArr[o] * h + bArr[o];
    out[o] = h;
  }
  if (t0 + T == L) out[(size_t)L * N + n] = h;  // hs[-1] duplicate
}

// ---------------------------------------------------------------------------
extern "C" void kernel_launch(void* const* d_in, const int* in_sizes, int n_in,
                              void* d_out, int out_size, void* d_ws, size_t ws_size,
                              hipStream_t stream) {
  const float*         u  = (const float*)d_in[0];          // (L,N)
  const float*         x  = (const float*)d_in[1];          // (N,)
  const unsigned char* dd = (const unsigned char*)d_in[2];  // (L,) bool
  const float*         Wz = (const float*)d_in[3];          // (N,N)
  const float*         bz = (const float*)d_in[4];          // (N,)
  const float*         Wh = (const float*)d_in[5];          // (N,N)
  const float*         bh = (const float*)d_in[6];          // (N,)

  const int N = in_sizes[1];   // 2048
  const int L = in_sizes[2];   // 8192
  const int C = 64;            // scan chunks
  const int T = L / C;         // 128 steps per chunk

  char* ws = (char*)d_ws;
  size_t off = 0;
  auto walloc = [&](size_t bytes) -> char* {
    char* p = ws + off;
    off = (off + bytes + 255) & ~(size_t)255;
    return p;
  };
  unsigned short* ubf  = (unsigned short*)walloc((size_t)L * N * 2);  // 32 MB
  unsigned short* wzbf = (unsigned short*)walloc((size_t)N * N * 2);  //  8 MB
  unsigned short* whbf = (unsigned short*)walloc((size_t)N * N * 2);  //  8 MB
  float* aArr = (float*)walloc((size_t)L * N * 4);                    // 64 MB
  float* bArr = (float*)walloc((size_t)L * N * 4);                    // 64 MB
  float* Ac   = (float*)walloc((size_t)C * N * 4);
  float* Bc   = (float*)walloc((size_t)C * N * 4);
  float* Hc   = (float*)walloc((size_t)C * N * 4);
  (void)ws_size; (void)n_in; (void)out_size;

  cvt_bf16_kernel<<<2048, 256, 0, stream>>>(u,  ubf,  (size_t)L * N);
  cvt_bf16_kernel<<<1024, 256, 0, stream>>>(Wz, wzbf, (size_t)N * N);
  cvt_bf16_kernel<<<1024, 256, 0, stream>>>(Wh, whbf, (size_t)N * N);

  dim3 ggrid(N / 128, L / 64);   // (16, 128) workgroups
  gemm_gate_kernel<<<ggrid, 256, 0, stream>>>(ubf, wzbf, whbf, bz, bh, dd,
                                              aArr, bArr, N);

  dim3 sgrid(N / 256, C);
  chunk_reduce_kernel<<<sgrid, 256, 0, stream>>>(aArr, bArr, Ac, Bc, N, T);
  carry_scan_kernel<<<N / 256, 256, 0, stream>>>(Ac, Bc, x, Hc, N, C);
  chunk_apply_kernel<<<sgrid, 256, 0, stream>>>(aArr, bArr, Hc, (float*)d_out,
                                                N, T, L);
}
